// RENet_1726576854247
// MI455X (gfx1250) — matrix operand order repaired and checked by reading, not compiled, not run on hardware
//
#include <hip/hip_runtime.h>

typedef __attribute__((ext_vector_type(16))) _Float16 v16h;
typedef __attribute__((ext_vector_type(8)))  float    v8f;

#define EPS 1e-5f

// ---------------------------------------------------------------------------
// 1) spatial self-correlation (5x5) + L2 norm  -> ssl (B,25,H,W)
// ---------------------------------------------------------------------------
__global__ __launch_bounds__(128) void k_ssl(const float* __restrict__ x,
                                             float* __restrict__ ssl) {
  int pix = blockIdx.x;                // b*256 + hw
  int b = pix >> 8, hw = pix & 255;
  int h = hw >> 4, w = hw & 15;
  const float* xb = x + (long long)b * 640 * 256;
  float acc[25];
#pragma unroll
  for (int s = 0; s < 25; ++s) acc[s] = 0.f;
  for (int c = threadIdx.x; c < 640; c += 128) {
    float xc = xb[c * 256 + hw];
#pragma unroll
    for (int di = 0; di < 5; ++di) {
      int hh = h + di - 2;
#pragma unroll
      for (int dj = 0; dj < 5; ++dj) {
        int ww = w + dj - 2;
        float xn = (hh >= 0 && hh < 16 && ww >= 0 && ww < 16)
                       ? xb[c * 256 + hh * 16 + ww] : 0.f;
        acc[di * 5 + dj] += xc * xn;
      }
    }
  }
  __shared__ float red[25][4];
  __shared__ float rn;
  int lane = threadIdx.x & 31, wv = threadIdx.x >> 5;
#pragma unroll
  for (int s = 0; s < 25; ++s) {
    float v = acc[s];
    for (int off = 16; off; off >>= 1) v += __shfl_down(v, off, 32);
    if (lane == 0) red[s][wv] = v;
  }
  __syncthreads();
  if (threadIdx.x < 25) {
    float v = red[threadIdx.x][0] + red[threadIdx.x][1] +
              red[threadIdx.x][2] + red[threadIdx.x][3];
    red[threadIdx.x][0] = v;
  }
  __syncthreads();
  if (threadIdx.x == 0) {
    float ss = 1e-6f;
    for (int s = 0; s < 25; ++s) ss += red[s][0] * red[s][0];
    rn = rsqrtf(ss);
  }
  __syncthreads();
  if (threadIdx.x < 25)
    ssl[((long long)b * 25 + threadIdx.x) * 256 + hw] = red[threadIdx.x][0] * rn;
}

// ---------------------------------------------------------------------------
// 2) local self-attention core -> attn (B,64,HW)  (pre-BN)
// ---------------------------------------------------------------------------
__global__ __launch_bounds__(64) void k_lsa(const float* __restrict__ ssl,
                                            const float* __restrict__ wq,
                                            const float* __restrict__ wk,
                                            const float* __restrict__ wv,
                                            const float* __restrict__ rel_h,
                                            const float* __restrict__ rel_w,
                                            float* __restrict__ attn) {
  int pix = blockIdx.x;
  int b = pix >> 8, hw = pix & 255, h = hw >> 4, w = hw & 15;
  __shared__ float s[9][25];
  const float* sb = ssl + (long long)b * 25 * 256;
  for (int t = threadIdx.x; t < 9 * 25; t += 64) {
    int nb = t / 25, c = t % 25;
    int hh = h + nb / 3 - 1, ww = w + nb % 3 - 1;
    s[nb][c] = (hh >= 0 && hh < 16 && ww >= 0 && ww < 16)
                   ? sb[c * 256 + hh * 16 + ww] : 0.f;
  }
  __syncthreads();
  int o = threadIdx.x;
  float q = 0.f;
#pragma unroll
  for (int c = 0; c < 25; ++c) q += wq[o * 25 + c] * s[4][c];
  float lg[9], vv[9], mx = -1e30f;
#pragma unroll
  for (int t = 0; t < 9; ++t) {
    float kk = 0.f, vl = 0.f;
#pragma unroll
    for (int c = 0; c < 25; ++c) {
      kk += wk[o * 25 + c] * s[t][c];
      vl += wv[o * 25 + c] * s[t][c];
    }
    kk += (o < 32) ? rel_h[o * 3 + t / 3] : rel_w[(o - 32) * 3 + t % 3];
    lg[t] = q * kk;
    vv[t] = vl;
    mx = fmaxf(mx, lg[t]);
  }
  float se = 0.f, out = 0.f;
#pragma unroll
  for (int t = 0; t < 9; ++t) {
    float e = __expf(lg[t] - mx);
    se += e;
    out += e * vv[t];
  }
  attn[((long long)b * 64 + o) * 256 + hw] = out / se;
}

// ---------------------------------------------------------------------------
// Deterministic two-pass BN statistics (per channel)
//   element address = c*chStride + outer*oStride + inner
// ---------------------------------------------------------------------------
template <typename T>
__global__ __launch_bounds__(256) void k_stats_part(const T* __restrict__ x,
                                                    long long chStride, int nOuter,
                                                    long long oStride, int nInner,
                                                    float* __restrict__ partial) {
  int c = blockIdx.x, part = blockIdx.y, parts = gridDim.y;
  float s = 0.f, s2 = 0.f;
  for (int o = part; o < nOuter; o += parts) {
    const T* row = x + (long long)c * chStride + (long long)o * oStride;
    for (int in = threadIdx.x; in < nInner; in += 256) {
      float v = (float)row[in];
      s += v; s2 += v * v;
    }
  }
  __shared__ float rs[256], rs2[256];
  rs[threadIdx.x] = s; rs2[threadIdx.x] = s2;
  __syncthreads();
  for (int off = 128; off; off >>= 1) {
    if (threadIdx.x < off) {
      rs[threadIdx.x] += rs[threadIdx.x + off];
      rs2[threadIdx.x] += rs2[threadIdx.x + off];
    }
    __syncthreads();
  }
  if (threadIdx.x == 0) {
    partial[(c * parts + part) * 2 + 0] = rs[0];
    partial[(c * parts + part) * 2 + 1] = rs2[0];
  }
}

__global__ void k_stats_fin(const float* __restrict__ partial, int parts,
                            float invN, float* __restrict__ mv) {
  int c = blockIdx.x;
  if (threadIdx.x == 0) {
    float s = 0.f, s2 = 0.f;
    for (int p = 0; p < parts; ++p) {
      s += partial[(c * parts + p) * 2 + 0];
      s2 += partial[(c * parts + p) * 2 + 1];
    }
    float m = s * invN;
    mv[2 * c + 0] = m;
    mv[2 * c + 1] = fmaxf(0.f, s2 * invN - m * m);
  }
}

// ---------------------------------------------------------------------------
// 3) BN+ReLU then 64x64 aggregation matmul -> out2 (B,64,HW)
// ---------------------------------------------------------------------------
__global__ __launch_bounds__(64) void k_agg(const float* __restrict__ attn,
                                            const float* __restrict__ mv,
                                            const float* __restrict__ g1,
                                            const float* __restrict__ b1,
                                            const float* __restrict__ aggw,
                                            float* __restrict__ out2) {
  int pix = blockIdx.x;
  int b = pix >> 8, hw = pix & 255;
  __shared__ float t[64];
  int o = threadIdx.x;
  float xv = attn[((long long)b * 64 + o) * 256 + hw];
  t[o] = fmaxf(0.f, (xv - mv[2 * o]) * rsqrtf(mv[2 * o + 1] + EPS) * g1[o] + b1[o]);
  __syncthreads();
  float acc = 0.f;
#pragma unroll
  for (int c = 0; c < 64; ++c) acc += aggw[o * 64 + c] * t[c];
  out2[((long long)b * 64 + o) * 256 + hw] = acc;
}

// 4) BN + L2 norm -> f, stored fp16 pixel-major (B, 256 pix, 64 ch)
__global__ __launch_bounds__(64) void k_fnorm(const float* __restrict__ out2,
                                              const float* __restrict__ mv,
                                              const float* __restrict__ g2,
                                              const float* __restrict__ b2,
                                              _Float16* __restrict__ fh) {
  int pix = blockIdx.x;
  int b = pix >> 8, hw = pix & 255;
  int o = threadIdx.x;
  __shared__ float t[64];
  __shared__ float rn;
  float xv = out2[((long long)b * 64 + o) * 256 + hw];
  float y = (xv - mv[2 * o]) * rsqrtf(mv[2 * o + 1] + EPS) * g2[o] + b2[o];
  t[o] = y;
  __syncthreads();
  if (o == 0) {
    float ss = 1e-6f;
    for (int c = 0; c < 64; ++c) ss += t[c] * t[c];
    rn = rsqrtf(ss);
  }
  __syncthreads();
  fh[((long long)b * 256 + hw) * 64 + o] = (_Float16)(y * rn);
}

// ---------------------------------------------------------------------------
// 5) corr[b,p,q] = sum_c f[c,p]*f[c,q] : WMMA fp16 GEMM, one 16x16 tile/wave
// ---------------------------------------------------------------------------
__global__ __launch_bounds__(256) void k_corr(const _Float16* __restrict__ fh,
                                              float* __restrict__ corr) {
  int wave = threadIdx.x >> 5, lane = threadIdx.x & 31;
  int tile = blockIdx.x * 8 + wave;       // 4096 tiles total
  int b = tile >> 8, tt = tile & 255;
  int p0 = (tt >> 4) << 4, q0 = (tt & 15) << 4;
  const _Float16* fb = fh + (long long)b * 256 * 64;
  int m = lane & 15, hi = lane >> 4;
  v8f acc = {};
#pragma unroll
  for (int kb = 0; kb < 64; kb += 32) {
    v16h A, Bv;
#pragma unroll
    for (int e = 0; e < 16; ++e) {
      int j = e >> 1;
      // A layout (ISA 7.12.2, 16-bit A 16x32)
      int ka = kb + ((j >> 2) << 4) + (hi << 3) + ((j & 3) << 1) + (e & 1);
      A[e] = fb[(p0 + m) * 64 + ka];
      // B layout (same K order as A: ISA 7.12.2 16-bit layout)
      int kB = kb + ((e >> 3) << 4) + (hi << 3) + (e & 7);
      Bv[e] = fb[(q0 + m) * 64 + kB];
    }
    acc = __builtin_amdgcn_wmma_f32_16x16x32_f16(false, A, false, Bv,
                                                 (short)0, acc, false, false);
  }
  float* cb = corr + (long long)b * 65536;
#pragma unroll
  for (int r = 0; r < 8; ++r)
    cb[(p0 + r + hi * 8) * 256 + q0 + m] = acc[r];
}

// ---------------------------------------------------------------------------
// 6) L1 conv over (u,v)=i axis, 1ch 3x3 -> t1
// ---------------------------------------------------------------------------
__global__ __launch_bounds__(256) void k_cuv1(const float* __restrict__ corr,
                                              const float* __restrict__ w,
                                              float* __restrict__ t1) {
  int bi = blockIdx.x;
  int b = bi >> 8, i = bi & 255, u = i >> 4, v = i & 15;
  int j = threadIdx.x;
  float acc = 0.f;
#pragma unroll
  for (int du = -1; du <= 1; ++du)
#pragma unroll
    for (int dv = -1; dv <= 1; ++dv) {
      int uu = u + du, vv = v + dv;
      if (uu >= 0 && uu < 16 && vv >= 0 && vv < 16)
        acc += w[(du + 1) * 3 + dv + 1] *
               corr[((long long)b * 256 + uu * 16 + vv) * 256 + j];
    }
  t1[((long long)b * 256 + i) * 256 + j] = acc;
}

// 7) L1: BN+ReLU then conv over (h,w)=j axis, 1ch 3x3 -> t2
__global__ __launch_bounds__(256) void k_chw1(const float* __restrict__ t1,
                                              const float* __restrict__ mv,
                                              const float* __restrict__ gp,
                                              const float* __restrict__ bp,
                                              const float* __restrict__ w,
                                              float* __restrict__ t2) {
  int bi = blockIdx.x;
  int b = bi >> 8, i = bi & 255;
  int j = threadIdx.x;
  __shared__ float r[256];
  float x = t1[((long long)b * 256 + i) * 256 + j];
  r[j] = fmaxf(0.f, (x - mv[0]) * rsqrtf(mv[1] + EPS) * gp[0] + bp[0]);
  __syncthreads();
  int hh = j >> 4, ww = j & 15;
  float acc = 0.f;
#pragma unroll
  for (int dh = -1; dh <= 1; ++dh)
#pragma unroll
    for (int dw = -1; dw <= 1; ++dw) {
      int h2 = hh + dh, w2 = ww + dw;
      if (h2 >= 0 && h2 < 16 && w2 >= 0 && w2 < 16)
        acc += w[(dh + 1) * 3 + dw + 1] * r[h2 * 16 + w2];
    }
  t2[((long long)b * 256 + i) * 256 + j] = acc;
}

// 8) fold L1 tail (BN -> 1->16 proj -> BN16) into per-channel affine (exact:
//    BN output has zero mean, per-channel var scales with pw^2)
__global__ void k_l1coef(const float* __restrict__ mv2,
                         const float* __restrict__ g1p, const float* __restrict__ b1p,
                         const float* __restrict__ pw, const float* __restrict__ pg,
                         const float* __restrict__ pb, float* __restrict__ coef) {
  int p = threadIdx.x;
  if (p >= 16) return;
  float m2 = mv2[0], v2 = mv2[1];
  float inv2 = rsqrtf(v2 + EPS);
  float vz = v2 / (v2 + EPS);
  float g1 = g1p[0];
  float pwp = pw[p];
  float invp = rsqrtf(pwp * pwp * g1 * g1 * vz + EPS);
  float A = pwp * g1 * pg[p] * invp * inv2;
  coef[p * 2 + 0] = A;
  coef[p * 2 + 1] = pb[p] - A * m2;
}

// pack conv weights (16,16,3,3) fp32 -> fp16 [c][K=160] (K = p*9+tap, zero pad)
__global__ void k_packw(const float* __restrict__ w, _Float16* __restrict__ wp) {
  int t = threadIdx.x + blockIdx.x * blockDim.x;
  if (t >= 16 * 160) return;
  int k = t % 160;
  wp[t] = (k < 144) ? (_Float16)w[(t / 160) * 144 + k] : (_Float16)0.f;
}

// ---------------------------------------------------------------------------
// implicit-GEMM 16->16 3x3 conv tile via WMMA: M=16 out ch, K=160, N=16 cols
// u: LDS [16 ch][256 pos], wp: global [16][160]
// ---------------------------------------------------------------------------
__device__ __forceinline__ v8f conv16_tile(const _Float16* __restrict__ wp,
                                           const _Float16* u, int n0, int lane) {
  int m = lane & 15, hi = lane >> 4;
  v8f acc = {};
#pragma unroll
  for (int kb = 0; kb < 160; kb += 32) {
    v16h A, Bv;
#pragma unroll
    for (int e = 0; e < 16; ++e) {
      int j = e >> 1;
      int ka = kb + ((j >> 2) << 4) + (hi << 3) + ((j & 3) << 1) + (e & 1);
      A[e] = wp[m * 160 + ka];
      int kB = kb + ((e >> 3) << 4) + (hi << 3) + (e & 7);
      _Float16 bval = (_Float16)0.f;
      if (kB < 144) {
        int p = kB / 9, tap = kB - p * 9;
        int ii = n0 + m;                       // column position (conv axis)
        int uu = (ii >> 4) + tap / 3 - 1;
        int vv = (ii & 15) + tap % 3 - 1;
        if (uu >= 0 && uu < 16 && vv >= 0 && vv < 16)
          bval = u[p * 256 + uu * 16 + vv];
      }
      Bv[e] = bval;
    }
    acc = __builtin_amdgcn_wmma_f32_16x16x32_f16(false, A, false, Bv,
                                                 (short)0, acc, false, false);
  }
  return acc;
}

// 9) L2 conv over (u,v): expand 1ch->16ch on the fly, WMMA conv -> t3 (b,i,c,j) f16
__global__ __launch_bounds__(256) void k_cuv2(const float* __restrict__ t2,
                                              const float* __restrict__ coef,
                                              const _Float16* __restrict__ wp,
                                              _Float16* __restrict__ t3) {
  __shared__ _Float16 u[16 * 256];
  int bj = blockIdx.x;
  int b = bj >> 8, j = bj & 255;
  int i = threadIdx.x;
  float x = t2[((long long)b * 256 + i) * 256 + j];
#pragma unroll
  for (int p = 0; p < 16; ++p)
    u[p * 256 + i] = (_Float16)fmaxf(0.f, coef[2 * p] * x + coef[2 * p + 1]);
  __syncthreads();
  int wave = threadIdx.x >> 5, lane = threadIdx.x & 31;
  for (int tile = wave; tile < 16; tile += 8) {
    int n0 = tile * 16;
    v8f acc = conv16_tile(wp, u, n0, lane);
    int icol = n0 + (lane & 15), hi = lane >> 4;
#pragma unroll
    for (int r = 0; r < 8; ++r)
      t3[(((long long)b * 256 + icol) * 16 + (r + hi * 8)) * 256 + j] =
          (_Float16)acc[r];
  }
}

// 10) L2: BN16+ReLU then conv over (h,w) via WMMA -> t4 (b,i,c,j) f16
__global__ __launch_bounds__(256) void k_chw2(const _Float16* __restrict__ t3,
                                              const float* __restrict__ mv,
                                              const float* __restrict__ g,
                                              const float* __restrict__ bb,
                                              const _Float16* __restrict__ wp,
                                              _Float16* __restrict__ t4) {
  __shared__ _Float16 u[16 * 256];
  int bi = blockIdx.x;
  int b = bi >> 8, i = bi & 255;
  int j = threadIdx.x;
  const _Float16* src = t3 + ((long long)b * 256 + i) * 16 * 256;
#pragma unroll
  for (int p = 0; p < 16; ++p) {
    float x = (float)src[p * 256 + j];
    float y = fmaxf(0.f, (x - mv[2 * p]) * rsqrtf(mv[2 * p + 1] + EPS) * g[p] + bb[p]);
    u[p * 256 + j] = (_Float16)y;
  }
  __syncthreads();
  int wave = threadIdx.x >> 5, lane = threadIdx.x & 31;
  _Float16* dst = t4 + ((long long)b * 256 + i) * 16 * 256;
  for (int tile = wave; tile < 16; tile += 8) {
    int n0 = tile * 16;
    v8f acc = conv16_tile(wp, u, n0, lane);
    int jc = n0 + (lane & 15), hi = lane >> 4;
#pragma unroll
    for (int r = 0; r < 8; ++r)
      dst[(r + hi * 8) * 256 + jc] = (_Float16)acc[r];
  }
}

// 11) BN16 + 16->1 projection -> o (b,i,j) fp32
__global__ __launch_bounds__(256) void k_proj2(const _Float16* __restrict__ t4,
                                               const float* __restrict__ mv,
                                               const float* __restrict__ g,
                                               const float* __restrict__ bb,
                                               const float* __restrict__ pw,
                                               float* __restrict__ o) {
  int bi = blockIdx.x;
  int b = bi >> 8, i = bi & 255;
  int j = threadIdx.x;
  const _Float16* src = t4 + ((long long)b * 256 + i) * 16 * 256;
  float acc = 0.f;
#pragma unroll
  for (int c = 0; c < 16; ++c) {
    float x = (float)src[c * 256 + j];
    acc += pw[c] * ((x - mv[2 * c]) * rsqrtf(mv[2 * c + 1] + EPS) * g[c] + bb[c]);
  }
  o[((long long)b * 256 + i) * 256 + j] = acc;
}

// 12) final BN(1ch) + symmetric sum: out = ohat + ohat^T
__global__ __launch_bounds__(256) void k_final(const float* __restrict__ o,
                                               const float* __restrict__ mv,
                                               const float* __restrict__ g,
                                               const float* __restrict__ bb,
                                               float* __restrict__ out) {
  int bi = blockIdx.x;
  int b = bi >> 8, i = bi & 255;
  int j = threadIdx.x;
  float m = mv[0], inv = rsqrtf(mv[1] + EPS);
  float a = (o[((long long)b * 256 + i) * 256 + j] - m) * inv * g[0] + bb[0];
  float c = (o[((long long)b * 256 + j) * 256 + i] - m) * inv * g[0] + bb[0];
  out[((long long)b * 256 + i) * 256 + j] = a + c;
}

// ---------------------------------------------------------------------------
extern "C" void kernel_launch(void* const* d_in, const int* in_sizes, int n_in,
                              void* d_out, int out_size, void* d_ws, size_t ws_size,
                              hipStream_t stream) {
  (void)in_sizes; (void)n_in; (void)out_size; (void)ws_size;
  const float* x      = (const float*)d_in[0];
  const float* wq     = (const float*)d_in[1];
  const float* wk     = (const float*)d_in[2];
  const float* wv     = (const float*)d_in[3];
  const float* rel_h  = (const float*)d_in[4];
  const float* rel_w  = (const float*)d_in[5];
  const float* agg_g1 = (const float*)d_in[6];
  const float* agg_b1 = (const float*)d_in[7];
  const float* agg_w  = (const float*)d_in[8];
  const float* agg_g2 = (const float*)d_in[9];
  const float* agg_b2 = (const float*)d_in[10];
  const float* l1_w2  = (const float*)d_in[11];
  const float* l1_g2  = (const float*)d_in[12];
  const float* l1_b2  = (const float*)d_in[13];
  const float* l1_w1  = (const float*)d_in[14];
  const float* l1_g1  = (const float*)d_in[15];
  const float* l1_b1  = (const float*)d_in[16];
  const float* l1_pw  = (const float*)d_in[17];
  const float* l1_pg  = (const float*)d_in[18];
  const float* l1_pb  = (const float*)d_in[19];
  const float* l2_w2  = (const float*)d_in[20];
  const float* l2_g2  = (const float*)d_in[21];
  const float* l2_b2  = (const float*)d_in[22];
  const float* l2_w1  = (const float*)d_in[23];
  const float* l2_g1  = (const float*)d_in[24];
  const float* l2_b1  = (const float*)d_in[25];
  const float* l2_pw  = (const float*)d_in[26];
  const float* l2_pg  = (const float*)d_in[27];
  const float* l2_pb  = (const float*)d_in[28];
  float* out = (float*)d_out;

  char* ws = (char*)d_ws;
  size_t off = 0;
  auto alloc = [&](size_t bytes) {
    size_t o = off;
    off = (off + bytes + 255) & ~(size_t)255;
    return o;
  };
  float*    ssl    = (float*)(ws + alloc(102400 * 4));
  float*    attn   = (float*)(ws + alloc(262144 * 4));
  float*    out2   = (float*)(ws + alloc(262144 * 4));
  _Float16* fh     = (_Float16*)(ws + alloc(262144 * 2));
  float*    corr   = (float*)(ws + alloc(1048576 * 4));
  float*    t1     = (float*)(ws + alloc(1048576 * 4));
  float*    t2     = (float*)(ws + alloc(1048576 * 4));
  _Float16* t3     = (_Float16*)(ws + alloc((size_t)16777216 * 2));
  _Float16* t4     = (_Float16*)(ws + alloc((size_t)16777216 * 2));
  float*    obuf   = (float*)(ws + alloc(1048576 * 4));
  _Float16* wp2    = (_Float16*)(ws + alloc(16 * 160 * 2));
  _Float16* wp1    = (_Float16*)(ws + alloc(16 * 160 * 2));
  float*    part   = (float*)(ws + alloc(64 * 64 * 2 * 4));
  float*    mvreg  = (float*)(ws + alloc(512 * 4));
  float* mv_attn = mvreg + 0;    // 128
  float* mv_o2   = mvreg + 128;  // 128
  float* mv_t1   = mvreg + 256;  // 2
  float* mv_t2   = mvreg + 260;  // 2
  float* mv_t3   = mvreg + 264;  // 32
  float* mv_t4   = mvreg + 300;  // 32
  float* mv_o    = mvreg + 336;  // 2
  float* coef    = mvreg + 344;  // 32

  const float invPix = 1.f / 4096.f;    // per-channel N for (B,64,HW)
  const float inv6D  = 1.f / 1048576.f; // per-channel N for 6-D tensors

  // stage 1-2
  k_ssl<<<4096, 128, 0, stream>>>(x, ssl);
  k_lsa<<<4096, 64, 0, stream>>>(ssl, wq, wk, wv, rel_h, rel_w, attn);
  // BN stats of attn (C=64)
  k_stats_part<float><<<dim3(64, 8), 256, 0, stream>>>(attn, 256LL, 16, 16384LL, 256, part);
  k_stats_fin<<<64, 32, 0, stream>>>(part, 8, invPix, mv_attn);
  // stage 3-4
  k_agg<<<4096, 64, 0, stream>>>(attn, mv_attn, agg_g1, agg_b1, agg_w, out2);
  k_stats_part<float><<<dim3(64, 8), 256, 0, stream>>>(out2, 256LL, 16, 16384LL, 256, part);
  k_stats_fin<<<64, 32, 0, stream>>>(part, 8, invPix, mv_o2);
  k_fnorm<<<4096, 64, 0, stream>>>(out2, mv_o2, agg_g2, agg_b2, fh);
  // stage 5: correlation GEMM (WMMA)
  k_corr<<<512, 256, 0, stream>>>(fh, corr);
  // L1
  k_cuv1<<<4096, 256, 0, stream>>>(corr, l1_w2, t1);
  k_stats_part<float><<<dim3(1, 64), 256, 0, stream>>>(t1, 0LL, 4096, 256LL, 256, part);
  k_stats_fin<<<1, 32, 0, stream>>>(part, 64, inv6D, mv_t1);
  k_chw1<<<4096, 256, 0, stream>>>(t1, mv_t1, l1_g2, l1_b2, l1_w1, t2);
  k_stats_part<float><<<dim3(1, 64), 256, 0, stream>>>(t2, 0LL, 4096, 256LL, 256, part);
  k_stats_fin<<<1, 32, 0, stream>>>(part, 64, inv6D, mv_t2);
  k_l1coef<<<1, 16, 0, stream>>>(mv_t2, l1_g1, l1_b1, l1_pw, l1_pg, l1_pb, coef);
  // L2 (WMMA implicit-GEMM convs)
  k_packw<<<10, 256, 0, stream>>>(l2_w2, wp2);
  k_packw<<<10, 256, 0, stream>>>(l2_w1, wp1);
  k_cuv2<<<4096, 256, 0, stream>>>(t2, coef, wp2, t3);
  k_stats_part<_Float16><<<dim3(16, 64), 256, 0, stream>>>(t3, 256LL, 4096, 4096LL, 256, part);
  k_stats_fin<<<16, 32, 0, stream>>>(part, 64, inv6D, mv_t3);
  k_chw2<<<4096, 256, 0, stream>>>(t3, mv_t3, l2_g2, l2_b2, wp1, t4);
  k_stats_part<_Float16><<<dim3(16, 64), 256, 0, stream>>>(t4, 256LL, 4096, 4096LL, 256, part);
  k_stats_fin<<<16, 32, 0, stream>>>(part, 64, inv6D, mv_t4);
  k_proj2<<<4096, 256, 0, stream>>>(t4, mv_t4, l2_g1, l2_b1, l2_pw, obuf);
  k_stats_part<float><<<dim3(1, 64), 256, 0, stream>>>(obuf, 0LL, 4096, 256LL, 256, part);
  k_stats_fin<<<1, 32, 0, stream>>>(part, 64, inv6D, mv_o);
  // final: branch(corr) + branch(corr)^T  (corr is symmetric -> one branch)
  k_final<<<4096, 256, 0, stream>>>(obuf, mv_o, l2_pg, l2_pb, out);
}
